// RnnSenderReinforce_31980326486493
// MI455X (gfx1250) — matrix operand order repaired and checked by reading, not compiled, not run on hardware
//
#include <hip/hip_runtime.h>

// Problem constants (from reference): B=512, IN=1024, H=E=512, V=2048, LEN=32
#define NB    512
#define NIN   1024
#define NH    512
#define NE    512
#define NV    2048
#define LEN   32

typedef __attribute__((ext_vector_type(16))) __bf16 v16bf;
typedef __attribute__((ext_vector_type(8)))  float  v8f;

__device__ __forceinline__ __bf16 f2bf(float f) {
    unsigned u = __builtin_bit_cast(unsigned, f);
    unsigned r = (u + 0x7FFFu + ((u >> 16) & 1u)) >> 16;   // RNE
    unsigned short s = (unsigned short)r;
    return __builtin_bit_cast(__bf16, s);
}

__device__ __forceinline__ float sigm(float x) { return 1.0f / (1.0f + expf(-x)); }

__device__ __forceinline__ unsigned hash3(unsigned a, unsigned b, unsigned c) {
    unsigned h = a * 0x9E3779B1u + b * 0x85EBCA77u + c * 0xC2B2AE3Du + 0x27220A95u;
    h ^= h >> 15; h *= 0x2C1B3C6Du;
    h ^= h >> 12; h *= 0x297A2D39u;
    h ^= h >> 15;
    return h;
}

__device__ __forceinline__ float wredmax(float v) {
    #pragma unroll
    for (int o = 16; o > 0; o >>= 1) v = fmaxf(v, __shfl_xor(v, o, 32));
    return v;
}
__device__ __forceinline__ float wredsum(float v) {
    #pragma unroll
    for (int o = 16; o > 0; o >>= 1) { float t = __shfl_xor(v, o, 32); v += t; }
    return v;
}

// inverse of A-fragment K mapping: kk (0..31) -> (half, j)
// half = (kk>>3)&1 ; j = (kk&7) + ((kk>>4)&1)*8
__device__ __forceinline__ void store_h_frag(__bf16* hF, int wave, int lane, int t, int i, float h) {
    int M    = i + ((lane >> 4) << 3);         // row in C/D layout
    int hcol = wave * 64 + t * 16 + (lane & 15);
    int kt   = hcol >> 5;
    int kk   = hcol & 31;
    int half = (kk >> 3) & 1;
    int j    = (kk & 7) + (((kk >> 4) & 1) << 3);
    hF[kt * 512 + (half * 16 + M) * 16 + j] = f2bf(h);
}

// ---------------- prep kernels: fp32 -> bf16 WMMA fragment layouts ----------------

// B operand fragments: W is [K,N] row-major; F is [NT][KT][32 lanes][16 j]
// lane<16: N=lane,   K = kt*32 + (j<8 ? j   : j+8)   (K 0..7,16..23)
// lane>=16:N=lane-16,K = kt*32 + (j<8 ? j+8 : j+16)  (K 8..15,24..31)
__global__ void prep_bfrag(const float* __restrict__ W, __bf16* __restrict__ F,
                           int K, int N, int total) {
    int idx = blockIdx.x * blockDim.x + threadIdx.x;
    if (idx >= total) return;
    int j    = idx & 15;
    int lane = (idx >> 4) & 31;
    int frag = idx >> 9;
    int KT   = K >> 5;
    int kt   = frag % KT;
    int nt   = frag / KT;
    int k    = kt * 32 + ((j >> 3) << 4) + ((lane >> 4) << 3) + (j & 7);
    int n    = nt * 16 + (lane & 15);
    F[idx] = f2bf(W[k * N + n]);
}

// A operand fragments: X is [rows,K] row-major; F is [rows/16][KT][32][16]
// lane<16: M=lane,    kk = j<8 ? j    : j+8   (K 0..7,16..23)
// lane>=16:M=lane-16, kk = j<8 ? j+8  : j+16  (K 8..15,24..31)
__global__ void prep_afrag(const float* __restrict__ X, __bf16* __restrict__ F,
                           int K, int total) {
    int idx = blockIdx.x * blockDim.x + threadIdx.x;
    if (idx >= total) return;
    int j    = idx & 15;
    int lane = (idx >> 4) & 31;
    int frag = idx >> 9;
    int KT   = K >> 5;
    int kt   = frag % KT;
    int blk  = frag / KT;
    int half = lane >> 4;
    int m    = lane & 15;
    int kk   = half == 0 ? (j < 8 ? j : j + 8) : (j < 8 ? j + 8 : j + 16);
    F[idx] = f2bf(X[(blk * 16 + m) * K + kt * 32 + kk]);
}

// ---------------- persistent LSTM sender kernel ----------------
// grid = 32 blocks (16 batch rows each), 256 threads = 8 waves.
// Wave w owns H-columns [w*64, w*64+64) across all 4 gates, and V-columns [w*256, w*256+256).
// Dynamic LDS: 128KB f32 logits + 16KB e-fragments + 16KB h-fragments = 160KB (CDNA5 WGP LDS).

#define WMMA_BF16(A, B, C) \
    __builtin_amdgcn_wmma_f32_16x16x32_bf16(false, (A), false, (B), (short)0, (C), false, false)

__global__ __launch_bounds__(256)
void lstm_sender(const float* __restrict__ agent_b,
                 const float* __restrict__ sos,
                 const float* __restrict__ embedding,
                 const float* __restrict__ b_ih,
                 const float* __restrict__ b_hh,
                 const float* __restrict__ out_b,
                 const __bf16* __restrict__ xF,
                 const __bf16* __restrict__ agwF,
                 const __bf16* __restrict__ wihF,
                 const __bf16* __restrict__ whhF,
                 const __bf16* __restrict__ outwF,
                 float* __restrict__ out) {
    extern __shared__ char smem[];
    float*  logitsBuf = (float*)smem;                              // [16][2048] f32
    __bf16* eFrag     = (__bf16*)(smem + 16 * NV * 4);             // [16 kt][32][16] bf16
    __bf16* hFrag     = (__bf16*)(smem + 16 * NV * 4 + 16384);     // [16 kt][32][16] bf16

    float* seqO   = out;
    float* probsO = out + (size_t)NB * (LEN + 1);
    float* logpO  = probsO + (size_t)NB * (LEN + 1) * NV;
    float* entO   = logpO + (size_t)NB * (LEN + 1);

    const int tid  = threadIdx.x;
    const int wave = tid >> 5;
    const int lane = tid & 31;
    const int nlo  = lane & 15;
    const int hi8  = (lane >> 4) << 3;
    const int blk  = blockIdx.x;

    // per-lane constant biases
    float gbias[16], obias[16], abias[4];
    #pragma unroll
    for (int g = 0; g < 4; ++g)
        #pragma unroll
        for (int t = 0; t < 4; ++t) {
            int n = g * NH + wave * 64 + t * 16 + nlo;
            gbias[g * 4 + t] = b_ih[n] + b_hh[n];
        }
    #pragma unroll
    for (int vt = 0; vt < 16; ++vt) obias[vt] = out_b[wave * 256 + vt * 16 + nlo];
    #pragma unroll
    for (int t = 0; t < 4; ++t) abias[t] = agent_b[wave * 64 + t * 16 + nlo];

    // e0 = broadcast(sos_embedding) into A-fragment layout
    for (int idx = tid; idx < 8192; idx += 256) {
        int j    = idx & 15;
        int ln   = (idx >> 4) & 31;
        int kt   = idx >> 9;
        int half = ln >> 4;
        int kk   = half == 0 ? (j < 8 ? j : j + 8) : (j < 8 ? j + 8 : j + 16);
        eFrag[idx] = f2bf(sos[kt * 32 + kk]);
    }

    // h0 = x @ agent_w + agent_b  (K = 1024 -> 32 k-tiles)
    v8f accH[4];
    #pragma unroll
    for (int t = 0; t < 4; ++t) accH[t] = (v8f){0, 0, 0, 0, 0, 0, 0, 0};
    for (int kt = 0; kt < 32; ++kt) {
        v16bf a = *(const v16bf*)&xF[(blk * 32 + kt) * 512 + lane * 16];
        #pragma unroll
        for (int t = 0; t < 4; ++t) {
            v16bf b = *(const v16bf*)&agwF[((wave * 4 + t) * 32 + kt) * 512 + lane * 16];
            accH[t] = WMMA_BF16(a, b, accH[t]);
        }
    }
    float cst[32];
    #pragma unroll
    for (int i = 0; i < 32; ++i) cst[i] = 0.0f;

    __syncthreads();
    #pragma unroll
    for (int t = 0; t < 4; ++t)
        #pragma unroll
        for (int i = 0; i < 8; ++i)
            store_h_frag(hFrag, wave, lane, t, i, accH[t][i] + abias[t]);
    __syncthreads();

    for (int step = 0; step < LEN; ++step) {
        // ---- gates = e@w_ih + h@w_hh : 16 n-tiles x 16 k-tiles x 2 ----
        v8f acc[16];
        #pragma unroll
        for (int i = 0; i < 16; ++i) acc[i] = (v8f){0, 0, 0, 0, 0, 0, 0, 0};
        for (int kt = 0; kt < 16; ++kt) {
            v16bf ae = *(const v16bf*)&eFrag[kt * 512 + lane * 16];
            v16bf ah = *(const v16bf*)&hFrag[kt * 512 + lane * 16];
            #pragma unroll
            for (int g = 0; g < 4; ++g)
                #pragma unroll
                for (int t = 0; t < 4; ++t) {
                    int nt = g * 32 + wave * 4 + t;
                    v16bf bi = *(const v16bf*)&wihF[(nt * 16 + kt) * 512 + lane * 16];
                    v16bf bh = *(const v16bf*)&whhF[(nt * 16 + kt) * 512 + lane * 16];
                    acc[g * 4 + t] = WMMA_BF16(ae, bi, acc[g * 4 + t]);
                    acc[g * 4 + t] = WMMA_BF16(ah, bh, acc[g * 4 + t]);
                }
        }
        __syncthreads();   // all waves done reading e/h fragments

        // ---- LSTM cell update (gate order i,f,g,o), write new h ----
        #pragma unroll
        for (int t = 0; t < 4; ++t)
            #pragma unroll
            for (int i = 0; i < 8; ++i) {
                float ig = sigm(acc[0 * 4 + t][i] + gbias[0 * 4 + t]);
                float fg = sigm(acc[1 * 4 + t][i] + gbias[1 * 4 + t]);
                float gg = tanhf(acc[2 * 4 + t][i] + gbias[2 * 4 + t]);
                float og = sigm(acc[3 * 4 + t][i] + gbias[3 * 4 + t]);
                float c  = fg * cst[t * 8 + i] + ig * gg;
                cst[t * 8 + i] = c;
                store_h_frag(hFrag, wave, lane, t, i, og * tanhf(c));
            }
        __syncthreads();

        // ---- logits = h @ out_w : 16 n-tiles x 16 k-tiles ----
        v8f accL[16];
        #pragma unroll
        for (int i = 0; i < 16; ++i) accL[i] = (v8f){0, 0, 0, 0, 0, 0, 0, 0};
        for (int kt = 0; kt < 16; ++kt) {
            v16bf ah = *(const v16bf*)&hFrag[kt * 512 + lane * 16];
            #pragma unroll
            for (int vt = 0; vt < 16; ++vt) {
                int nt = wave * 16 + vt;
                v16bf b = *(const v16bf*)&outwF[(nt * 16 + kt) * 512 + lane * 16];
                accL[vt] = WMMA_BF16(ah, b, accL[vt]);
            }
        }
        #pragma unroll
        for (int vt = 0; vt < 16; ++vt)
            #pragma unroll
            for (int i = 0; i < 8; ++i)
                logitsBuf[(i + hi8) * NV + wave * 256 + vt * 16 + nlo] = accL[vt][i] + obias[vt];
        __syncthreads();

        // ---- log-softmax / entropy / Gumbel-argmax sample / embed: wave per row ----
        #pragma unroll
        for (int rr = 0; rr < 2; ++rr) {
            int r  = wave * 2 + rr;
            int gb = blk * 16 + r;
            const float* lrow = &logitsBuf[r * NV];

            float mx = -1e30f;
            for (int q = 0; q < 64; ++q) mx = fmaxf(mx, lrow[q * 32 + lane]);
            mx = wredmax(mx);

            float s = 0.0f;
            for (int q = 0; q < 64; ++q) s += expf(lrow[q * 32 + lane] - mx);
            s = wredsum(s);
            float logZ = mx + logf(s);

            float entAcc = 0.0f, bestV = -1e30f;
            int   bestC  = 0;
            for (int q = 0; q < 64; ++q) {
                int   col = q * 32 + lane;
                float lp  = lrow[col] - logZ;
                float p   = expf(lp);
                entAcc += p * lp;
                probsO[((size_t)gb * (LEN + 1) + step) * NV + col] = p;
                unsigned hsh = hash3((unsigned)step, (unsigned)gb, (unsigned)col);
                float u   = fmaxf((float)(hsh >> 8) * (1.0f / 16777216.0f), 1e-20f);
                float key = lp - logf(-logf(u));
                if (key > bestV) { bestV = key; bestC = col; }
            }
            #pragma unroll
            for (int o = 16; o > 0; o >>= 1) {
                float ov = __shfl_xor(bestV, o, 32);
                int   oc = __shfl_xor(bestC, o, 32);
                float oe = __shfl_xor(entAcc, o, 32);
                entAcc += oe;
                if (ov > bestV || (ov == bestV && oc < bestC)) { bestV = ov; bestC = oc; }
            }
            int sym = bestC;
            if (lane == 0) {
                seqO[gb * (LEN + 1) + step]  = (float)sym;
                logpO[gb * (LEN + 1) + step] = lrow[sym] - logZ;
                entO[gb * (LEN + 1) + step]  = -entAcc;
            }
            // e <- embedding[sym], scattered into A-fragment layout for row r
            const float* erow = &embedding[(size_t)sym * NE];
            #pragma unroll
            for (int q = 0; q < 16; ++q) {
                int kk   = lane;
                int half = (kk >> 3) & 1;
                int j    = (kk & 7) + (((kk >> 4) & 1) << 3);
                eFrag[q * 512 + (half * 16 + r) * 16 + j] = f2bf(erow[q * 32 + lane]);
            }
        }
        __syncthreads();
    }

    // ---- EOS step (t = 32): seq/logp/ent = 0, probs = 1 ----
    for (int idx = tid; idx < 16 * NV; idx += 256) {
        int r = idx >> 11, col = idx & (NV - 1);
        probsO[((size_t)(blk * 16 + r) * (LEN + 1) + LEN) * NV + col] = 1.0f;
    }
    if (tid < 16) {
        int gb = blk * 16 + tid;
        seqO[gb * (LEN + 1) + LEN]  = 0.0f;
        logpO[gb * (LEN + 1) + LEN] = 0.0f;
        entO[gb * (LEN + 1) + LEN]  = 0.0f;
    }
}

// ---------------- launch ----------------
extern "C" void kernel_launch(void* const* d_in, const int* in_sizes, int n_in,
                              void* d_out, int out_size, void* d_ws, size_t ws_size,
                              hipStream_t stream) {
    (void)in_sizes; (void)n_in; (void)out_size; (void)ws_size;
    const float* x    = (const float*)d_in[0];   // [512,1024]
    const float* agw  = (const float*)d_in[1];   // [1024,512]
    const float* agb  = (const float*)d_in[2];   // [512]
    const float* sos  = (const float*)d_in[3];   // [512]
    const float* emb  = (const float*)d_in[4];   // [2048,512]
    const float* wih  = (const float*)d_in[5];   // [512,2048]
    const float* whh  = (const float*)d_in[6];   // [512,2048]
    const float* bih  = (const float*)d_in[7];   // [2048]
    const float* bhh  = (const float*)d_in[8];   // [2048]
    const float* outw = (const float*)d_in[9];   // [512,2048]
    const float* outb = (const float*)d_in[10];  // [2048]

    __bf16* ws    = (__bf16*)d_ws;
    __bf16* wihF  = ws;                  // 512*2048   = 1,048,576 bf16
    __bf16* whhF  = wihF + 1048576;
    __bf16* outwF = whhF + 1048576;
    __bf16* agwF  = outwF + 1048576;     // 1024*512   =   524,288
    __bf16* xF    = agwF + 524288;       // 512*1024   =   524,288

    prep_bfrag<<<1048576 / 256, 256, 0, stream>>>(wih,  wihF,  512, 2048, 1048576);
    prep_bfrag<<<1048576 / 256, 256, 0, stream>>>(whh,  whhF,  512, 2048, 1048576);
    prep_bfrag<<<1048576 / 256, 256, 0, stream>>>(outw, outwF, 512, 2048, 1048576);
    prep_bfrag<<<524288 / 256,  256, 0, stream>>>(agw,  agwF, 1024,  512,  524288);
    prep_afrag<<<524288 / 256,  256, 0, stream>>>(x,    xF,   1024,        524288);

    size_t lds = 16 * NV * 4 + 16384 + 16384;   // 160 KB dynamic LDS (CDNA5 WGP)
    lstm_sender<<<NB / 16, 256, lds, stream>>>(agb, sos, emb, bih, bhh, outb,
                                               xF, agwF, wihF, whhF, outwF,
                                               (float*)d_out);
}
